// NeuMIP_31482110280010
// MI455X (gfx1250) — matrix operand order repaired and checked by reading, not compiled, not run on hardware
//
#include <hip/hip_runtime.h>
#include <hip/hip_bf16.h>

// CDNA5 / gfx1250 NeuMIP forward: WMMA f16 MLPs + LDS-staged activations.

typedef __attribute__((ext_vector_type(16))) _Float16 v16h;
typedef __attribute__((ext_vector_type(8)))  _Float16 v8h;
typedef __attribute__((ext_vector_type(8)))  float    v8f;

#define RES 512
#define TCH 8

// Bilinear wrap-around fetch of 8 channels, per-thread (per-sample).
__device__ __forceinline__ void fetch2d_pt(const float* __restrict__ tex,
                                           float u01, float v01, float* out8) {
    const float Hm1 = (float)(RES - 1);
    float u = u01 * Hm1;
    float v = -v01 * Hm1;
    float u0f = floorf(u), v0f = floorf(v);
    float fu = u - u0f, fv = v - v0f;
    int u0 = ((int)u0f)      & (RES - 1);   // &(H-1) == mod H incl. negatives
    int v0 = ((int)v0f)      & (RES - 1);
    int u1 = ((int)ceilf(u)) & (RES - 1);
    int v1 = ((int)ceilf(v)) & (RES - 1);
    float w00 = (1.f - fu) * (1.f - fv);
    float w01 = fu * (1.f - fv);
    float w10 = (1.f - fu) * fv;
    float w11 = fu * fv;
    int i00 = v0 * RES + u0, i01 = v0 * RES + u1;
    int i10 = v1 * RES + u0, i11 = v1 * RES + u1;
#pragma unroll
    for (int t = 0; t < TCH; ++t) {
        const float* tp = tex + t * (RES * RES);
        out8[t] = tp[i00] * w00 + tp[i01] * w01 + tp[i10] * w10 + tp[i11] * w11;
    }
}

// One dense 32->32 layer with LeakyReLU(0.01) for this wave's 32 samples.
// src/dst: [256][32] f16 row-major activations in LDS. W: [out(N)][in(K)] f16.
// A frag (16x32 f16): lanes 0-15 hold row M=ln with K0-7,K16-23;
//                     lanes 16-31 hold same rows with K8-15,K24-31.
// B frag (32x16 f16): lanes 0-15 hold col N=ln with K0-7,K16-23; lanes 16-31 K8-15,K24-31.
// D (16x16 f32): lane = N col; VGPR r = row M (+8 for upper lanes).
__device__ __forceinline__ void dense32_layer(
    const _Float16 (*__restrict__ src)[32], _Float16 (*__restrict__ dst)[32],
    const _Float16 (*__restrict__ W)[32], const float* __restrict__ bias,
    int wave, int ln, int laneHi)
{
#pragma unroll
    for (int mt = 0; mt < 2; ++mt) {
        const int row = wave * 32 + mt * 16 + ln;
        union { v16h v; v8h h[2]; } A;
        A.h[0] = *(const v8h*)(&src[row][laneHi ? 8  : 0]);
        A.h[1] = *(const v8h*)(&src[row][laneHi ? 24 : 16]);
#pragma unroll
        for (int nt = 0; nt < 2; ++nt) {
            const int n = nt * 16 + ln;
            union { v16h v; v8h h[2]; } Bf;
            const v8h* bp = (const v8h*)(&W[n][laneHi ? 8 : 0]);
            Bf.h[0] = bp[0];
            Bf.h[1] = bp[2];
            v8f c = {};
            c = __builtin_amdgcn_wmma_f32_16x16x32_f16(
                    /*neg_a=*/false, A.v, /*neg_b=*/false, Bf.v,
                    /*c_mod=*/(short)0, c, /*reuse_a=*/false, /*reuse_b=*/false);
            const float bb = bias[n];
#pragma unroll
            for (int r = 0; r < 8; ++r) {
                float x = c[r] + bb;
                x = (x > 0.0f) ? x : 0.01f * x;           // LeakyReLU(0.01)
                dst[wave * 32 + mt * 16 + laneHi * 8 + r][n] = (_Float16)x;
            }
        }
    }
}

__global__ __launch_bounds__(256) void neumip_fwd_kernel(
    const float* __restrict__ uv,  const float* __restrict__ wo,
    const float* __restrict__ wi,
    const float* __restrict__ offtex, const float* __restrict__ rgbtex,
    const float* __restrict__ ow1, const float* __restrict__ ob1,
    const float* __restrict__ ow2, const float* __restrict__ ob2,
    const float* __restrict__ ow3, const float* __restrict__ ob3,
    const float* __restrict__ rw1, const float* __restrict__ rb1,
    const float* __restrict__ rw2, const float* __restrict__ rb2,
    const float* __restrict__ rw3, const float* __restrict__ rb3,
    float* __restrict__ out)
{
    __shared__ __align__(64) _Float16 sW1a[32][32];   // MLP1 L1 [out][in] padded
    __shared__ __align__(64) _Float16 sW2a[32][32];   // MLP1 L2
    __shared__ __align__(64) _Float16 sW1b[32][32];   // MLP2 L1 padded
    __shared__ __align__(64) _Float16 sW2b[32][32];   // MLP2 L2
    __shared__ float sB1a[32], sB2a[32], sB1b[32], sB2b[32];
    __shared__ float sW3a[32], sB3a[1];               // 32->1
    __shared__ float sW3b[3][32], sB3b[3];            // 32->3
    __shared__ __align__(64) _Float16 act0[256][32];  // ping
    __shared__ __align__(64) _Float16 act1[256][32];  // pong

    const int tid    = threadIdx.x;
    const int lane   = tid & 31;
    const int wave   = tid >> 5;
    const int laneHi = lane >> 4;   // 0 for lanes 0-15, 1 for 16-31
    const int ln     = lane & 15;
    const int s      = blockIdx.x * 256 + tid;

    // --- cooperative weight staging (only cross-wave sync in the kernel) ---
    for (int i = tid; i < 1024; i += 256) {
        const int r = i >> 5, c = i & 31;
        sW1a[r][c] = (_Float16)((c < 10) ? ow1[r * 10 + c] : 0.0f);
        sW2a[r][c] = (_Float16)ow2[i];
        sW1b[r][c] = (_Float16)((c < 12) ? rw1[r * 12 + c] : 0.0f);
        sW2b[r][c] = (_Float16)rw2[i];
    }
    if (tid < 32) {
        sB1a[tid] = ob1[tid]; sB2a[tid] = ob2[tid];
        sB1b[tid] = rb1[tid]; sB2b[tid] = rb2[tid];
        sW3a[tid] = ow3[tid];
    }
    if (tid < 96) sW3b[tid >> 5][tid & 31] = rw3[tid];
    if (tid < 3)  sB3b[tid] = rb3[tid];
    if (tid == 0) sB3a[0] = ob3[0];
    __syncthreads();

    // --- per-sample inputs ---
    const float uvx = uv[2 * s],  uvy = uv[2 * s + 1];
    const float wox = wo[2 * s],  woy = wo[2 * s + 1];
    const float wix = wi[2 * s],  wiy = wi[2 * s + 1];

    // --- offset-texture fetch + MLP1 feature row: [f_offset(8), wo(2), 0...] ---
    float f1[TCH];
    fetch2d_pt(offtex, uvx, uvy, f1);
#pragma unroll
    for (int k = 0; k < 8; ++k) act0[tid][k] = (_Float16)f1[k];
    act0[tid][8] = (_Float16)wox;
    act0[tid][9] = (_Float16)woy;
#pragma unroll
    for (int k = 10; k < 32; ++k) act0[tid][k] = (_Float16)0.0f;
    asm volatile("s_wait_dscnt 0" ::: "memory");  // rows visible to wave frag loads

    // --- MLP1 hidden layers via WMMA ---
    dense32_layer(act0, act1, sW1a, sB1a, wave, ln, laneHi);
    dense32_layer(act1, act0, sW2a, sB2a, wave, ln, laneHi);

    // --- MLP1 output (32->1), linear, per-thread ---
    float rsc = sB3a[0];
#pragma unroll
    for (int k = 0; k < 32; ++k) rsc += (float)act0[tid][k] * sW3a[k];

    const float denom = sqrtf(fmaxf(1.0f - (wox * wox + woy * woy), 0.36f));
    const float scale = rsc / denom;
    const float unx = scale * wox + uvx;
    const float uny = scale * woy + uvy;

    // --- rgb-texture fetch + MLP2 feature row: [wi(2), wo(2), f_rgb(8), 0...] ---
    float f2[TCH];
    fetch2d_pt(rgbtex, unx, uny, f2);
    act1[tid][0] = (_Float16)wix;
    act1[tid][1] = (_Float16)wiy;
    act1[tid][2] = (_Float16)wox;
    act1[tid][3] = (_Float16)woy;
#pragma unroll
    for (int k = 0; k < 8; ++k) act1[tid][4 + k] = (_Float16)f2[k];
#pragma unroll
    for (int k = 12; k < 32; ++k) act1[tid][k] = (_Float16)0.0f;
    asm volatile("s_wait_dscnt 0" ::: "memory");

    // --- MLP2 hidden layers via WMMA ---
    dense32_layer(act1, act0, sW1b, sB1b, wave, ln, laneHi);
    dense32_layer(act0, act1, sW2b, sB2b, wave, ln, laneHi);

    // --- MLP2 output (32->3), linear + ReLU, per-thread ---
#pragma unroll
    for (int k2 = 0; k2 < 3; ++k2) {
        float acc = sB3b[k2];
#pragma unroll
        for (int k = 0; k < 32; ++k) acc += (float)act1[tid][k] * sW3b[k2][k];
        out[3 * s + k2] = fmaxf(acc, 0.0f);
    }
}

extern "C" void kernel_launch(void* const* d_in, const int* in_sizes, int n_in,
                              void* d_out, int out_size, void* d_ws, size_t ws_size,
                              hipStream_t stream) {
    (void)n_in; (void)d_ws; (void)ws_size; (void)out_size;
    const float* uv     = (const float*)d_in[0];
    const float* wo     = (const float*)d_in[1];
    const float* wi     = (const float*)d_in[2];
    const float* offtex = (const float*)d_in[3];
    const float* rgbtex = (const float*)d_in[4];
    const float* ow1 = (const float*)d_in[5];
    const float* ob1 = (const float*)d_in[6];
    const float* ow2 = (const float*)d_in[7];
    const float* ob2 = (const float*)d_in[8];
    const float* ow3 = (const float*)d_in[9];
    const float* ob3 = (const float*)d_in[10];
    const float* rw1 = (const float*)d_in[11];
    const float* rb1 = (const float*)d_in[12];
    const float* rw2 = (const float*)d_in[13];
    const float* rb2 = (const float*)d_in[14];
    const float* rw3 = (const float*)d_in[15];
    const float* rb3 = (const float*)d_in[16];
    float* out = (float*)d_out;

    const int B = in_sizes[0] / 2;        // 4194304
    const int blocks = (B + 255) / 256;   // 16384, B is an exact multiple of 256

    neumip_fwd_kernel<<<blocks, 256, 0, stream>>>(
        uv, wo, wi, offtex, rgbtex,
        ow1, ob1, ow2, ob2, ow3, ob3,
        rw1, rb1, rw2, rb2, rw3, rb3, out);
}
